// RGLRU_74337293959630
// MI455X (gfx1250) — matrix operand order repaired and checked by reading, not compiled, not run on hardware
//
#include <hip/hip_runtime.h>

// ---------------------------------------------------------------------------
// RG-LRU (Griffin/Hawk recurrent gated LRU) for MI455X (gfx1250, wave32).
//   B=8, L=4096, W=2048, H=16, bw=128.
// One workgroup per (batch, head): 128 WGs, 256 threads (8 waves).
// Per chunk of T=64 timesteps:
//   1) stage x[64x128] f32 -> LDS (non-temporal b128 global loads)
//   2) both gate matmuls via v_wmma_f32_16x16x32_bf16 (f32 accum), fused
//      sigmoid / exp / sqrt elementwise (single-instruction rcp/sqrt paths),
//      write a[t][w], xhat[t][w] to LDS
//   3) 128 threads run the serial scan h = a*h + xhat (batched LDS loads)
//   4) 256 threads store y chunk with non-temporal b128 stores
// Weights are bf16-transposed into LDS once per block and stay resident.
// ---------------------------------------------------------------------------

typedef __attribute__((ext_vector_type(16))) __bf16 v16bf;
typedef __attribute__((ext_vector_type(8)))  float  v8f;
typedef __attribute__((ext_vector_type(4)))  float  v4f;   // native vector for NT ld/st

namespace {
constexpr int kB  = 8;
constexpr int kL  = 4096;
constexpr int kW  = 2048;
constexpr int kH  = 16;
constexpr int kBW = 128;   // head width
constexpr int kT  = 64;    // timesteps per chunk

// LDS strides (elements), padded for conflict-free banking:
constexpr int XST = 133;   // x chunk row stride (f32); A-frag reads (5m+k)%64 distinct
constexpr int AST = 132;   // a/xhat row stride (f32); multiple of 4 -> float4 rows
constexpr int KS  = 130;   // weight K-stride (bf16) per n-row; 65-dword stride

constexpr size_t kOffX   = 0;
constexpr size_t kOffA   = kOffX  + (size_t)kT * XST * 4;          // 34048
constexpr size_t kOffXN  = kOffA  + (size_t)kT * AST * 4;          // 67840
constexpr size_t kOffWI  = kOffXN + (size_t)kT * AST * 4;          // 101632
constexpr size_t kOffWA  = kOffWI + (size_t)kBW * KS * 2;          // 134912
constexpr size_t kOffBI  = kOffWA + (size_t)kBW * KS * 2;          // 168192
constexpr size_t kOffBA  = kOffBI + (size_t)kBW * 4;               // 168704
constexpr size_t kOffC   = kOffBA + (size_t)kBW * 4;               // 169216
constexpr size_t kLdsBytes = kOffC + (size_t)kBW * 4;              // 169728
}  // namespace

// sigmoid with single-instruction v_exp_f32 + v_rcp_f32 (gate math; ulp-level
// difference vs IEEE division is irrelevant post-sigmoid).
__device__ __forceinline__ float sigmoid_fast(float v) {
    return __builtin_amdgcn_rcpf(1.0f + __expf(-v));
}

__global__ __launch_bounds__(256, 1)
void rglru_fused_kernel(const float* __restrict__ x,
                        const float* __restrict__ a_param,
                        const float* __restrict__ ig_w,
                        const float* __restrict__ ig_b,
                        const float* __restrict__ ag_w,
                        const float* __restrict__ ag_b,
                        float* __restrict__ y,
                        float* __restrict__ last_h) {
    extern __shared__ char smem[];
    float*  x_l  = reinterpret_cast<float*>(smem + kOffX);
    float*  a_l  = reinterpret_cast<float*>(smem + kOffA);
    float*  xn_l = reinterpret_cast<float*>(smem + kOffXN);
    __bf16* w_l[2];
    w_l[0] = reinterpret_cast<__bf16*>(smem + kOffWI);
    w_l[1] = reinterpret_cast<__bf16*>(smem + kOffWA);
    float*  b_l[2];
    b_l[0] = reinterpret_cast<float*>(smem + kOffBI);
    b_l[1] = reinterpret_cast<float*>(smem + kOffBA);
    float*  c_l  = reinterpret_cast<float*>(smem + kOffC);

    const int tid  = threadIdx.x;
    const int lane = tid & 31;
    const int wv   = tid >> 5;          // wave 0..7
    const int trow = wv & 3;            // 16-row band of the chunk (4 bands)
    const int nhalf = wv >> 2;          // which 4 of the 8 N-tiles

    const int b = blockIdx.x >> 4;      // batch
    const int h = blockIdx.x & 15;      // head

    // ---- one-time: transpose weights to bf16 LDS [n][k] (K pairs contiguous)
    for (int e = tid; e < kBW * kBW; e += 256) {
        const int i = e >> 7;           // input dim (K)
        const int j = e & 127;          // output dim (N)
        const int g = (h * kBW + i) * kBW + j;
        w_l[0][j * KS + i] = (__bf16)ig_w[g];
        w_l[1][j * KS + i] = (__bf16)ag_w[g];
    }
    if (tid < kBW) {
        b_l[0][tid] = ig_b[h * kBW + tid];
        b_l[1][tid] = ag_b[h * kBW + tid];
        const float ap = a_param[h * kBW + tid];
        c_l[tid] = 8.0f * logf(1.0f + __expf(ap));   // 8*softplus(a_param)
    }
    // first in-loop barrier publishes weights before use

    // WMMA operand lane mapping (wave32; A and B both use contiguous k = 16*(lane/16) + e):
    const int m_a  = trow * 16 + (lane & 15);     // A row
    const int kb8  = (lane < 16) ? 0 : 8;         // unused; A K-base is kb16 as well
    const int kb16 = (lane < 16) ? 0 : 16;        // B K-base (32x16 bf16 B)
    const int nlane = lane & 15;
    const int mhi   = (lane >> 4) * 8;            // C: lanes 16-31 hold M+8

    const float* xg = x + (size_t)b * kL * kW + (size_t)h * kBW;
    float*       yg = y + (size_t)b * kL * kW + (size_t)h * kBW;

    float hreg = 0.0f;                            // recurrence state (tid<128)

    for (int chunk = 0; chunk < kL / kT; ++chunk) {
        const int Lbase = chunk * kT;

        // ---- phase 1: stage x chunk (coalesced, streaming/non-temporal)
        for (int idx = tid; idx < kT * 32; idx += 256) {
            const int r  = idx >> 5;
            const int c4 = (idx & 31) * 4;
            const v4f v = __builtin_nontemporal_load(
                reinterpret_cast<const v4f*>(xg + (size_t)(Lbase + r) * kW + c4));
            float* dst = x_l + r * XST + c4;
            dst[0] = v.x; dst[1] = v.y; dst[2] = v.z; dst[3] = v.w;
        }
        __syncthreads();

        // ---- phase 2: WMMA gate matmuls + fused elementwise
        // A fragments (shared by both gates): 16x32 bf16 per K-step.
        v16bf Af[4];
#pragma unroll
        for (int ks = 0; ks < 4; ++ks) {
#pragma unroll
            for (int e = 0; e < 16; ++e) {
                const int k = ks * 32 + kb16 + e;
                Af[ks][e] = (__bf16)x_l[m_a * XST + k];
            }
        }

#pragma unroll
        for (int ncq = 0; ncq < 4; ++ncq) {
            const int nc = nhalf * 4 + ncq;       // 16-col tile index
            v8f acc[2];
            acc[0] = {}; acc[1] = {};
#pragma unroll
            for (int g = 0; g < 2; ++g) {
                const __bf16* wb = w_l[g] + (nc * 16 + nlane) * KS;
#pragma unroll
                for (int ks = 0; ks < 4; ++ks) {
                    v16bf Bf;
#pragma unroll
                    for (int e = 0; e < 8; ++e) {
                        const int k = ks * 32 + kb16 + 2 * e;
                        Bf[2 * e]     = wb[k];
                        Bf[2 * e + 1] = wb[k + 1];
                    }
                    acc[g] = __builtin_amdgcn_wmma_f32_16x16x32_bf16(
                        false, Af[ks], false, Bf, (short)0, acc[g], false, false);
                }
            }
            // fused elementwise on the C tiles (both gates live in registers)
#pragma unroll
            for (int r = 0; r < 8; ++r) {
                const int tl  = trow * 16 + mhi + r;    // chunk-local t
                const int wch = nc * 16 + nlane;        // channel in head
                const float gx = sigmoid_fast(acc[0][r] + b_l[0][wch]);
                const float ga = sigmoid_fast(acc[1][r] + b_l[1][wch]);
                const float la = -c_l[wch] * ga;
                const float a  = __expf(la);
                const float a2 = a * a;                 // == exp(2*la)
                const float mult = (Lbase + tl == 0)
                                       ? 1.0f
                                       : __builtin_amdgcn_sqrtf(fmaxf(1.0f - a2, 0.0f));
                const float xv = x_l[tl * XST + wch];
                a_l[tl * AST + wch]  = a;
                xn_l[tl * AST + wch] = xv * gx * mult;
            }
        }
        __syncthreads();

        // ---- phase 3: serial scan over the chunk, batched to pipeline LDS
        if (tid < kBW) {
            for (int t0 = 0; t0 < kT; t0 += 8) {
                float av[8], xv[8];
#pragma unroll
                for (int u = 0; u < 8; ++u) {
                    av[u] = a_l[(t0 + u) * AST + tid];
                    xv[u] = xn_l[(t0 + u) * AST + tid];
                }
#pragma unroll
                for (int u = 0; u < 8; ++u) {
                    hreg = fmaf(av[u], hreg, xv[u]);
                    xn_l[(t0 + u) * AST + tid] = hreg;   // reuse buffer for y
                }
            }
        }
        __syncthreads();

        // ---- phase 4: store y chunk (b128 non-temporal; rows 16B aligned)
        for (int idx = tid; idx < kT * 32; idx += 256) {
            const int r  = idx >> 5;
            const int c4 = (idx & 31) * 4;
            const v4f v = *reinterpret_cast<const v4f*>(xn_l + r * AST + c4);
            __builtin_nontemporal_store(
                v, reinterpret_cast<v4f*>(yg + (size_t)(Lbase + r) * kW + c4));
        }
        // next iteration's first __syncthreads() orders these reads/stores
        // before phase-2 rewrites a_l/xn_l.
    }

    if (tid < kBW) {
        last_h[(size_t)b * kW + (size_t)h * kBW + tid] = hreg;
    }
}

extern "C" void kernel_launch(void* const* d_in, const int* in_sizes, int n_in,
                              void* d_out, int out_size, void* d_ws, size_t ws_size,
                              hipStream_t stream) {
    (void)in_sizes; (void)n_in; (void)out_size; (void)d_ws; (void)ws_size;
    const float* x  = (const float*)d_in[0];
    const float* ap = (const float*)d_in[1];
    const float* iw = (const float*)d_in[2];
    const float* ib = (const float*)d_in[3];
    const float* aw = (const float*)d_in[4];
    const float* ab = (const float*)d_in[5];

    float* y  = (float*)d_out;                       // [B, L, W]
    float* lh = y + (size_t)kB * kL * kW;            // [B, W]

    // Allow >64KB dynamic LDS (idempotent; not a stream op, capture-safe).
    (void)hipFuncSetAttribute(reinterpret_cast<const void*>(rglru_fused_kernel),
                              hipFuncAttributeMaxDynamicSharedMemorySize,
                              (int)kLdsBytes);

    rglru_fused_kernel<<<dim3(kB * kH), dim3(256), kLdsBytes, stream>>>(
        x, ap, iw, ib, aw, ab, y, lh);
}
